// ScaledDotProductAttention_88502096101603
// MI455X (gfx1250) — matrix operand order repaired and checked by reading, not compiled, not run on hardware
//
#include <hip/hip_runtime.h>
#include <hip/hip_bf16.h>

// ---------------------------------------------------------------------------
// Scaled dot-product attention for MI455X (gfx1250, wave32, WMMA).
// B=4 H=16 S=2048 D=128, fp32 in/out. Returns (attn_output, attn_weights).
//
// HBM-bound on the 1.07 GiB attn_weights write (~60us floor @ 23.3 TB/s):
//  - per-wave 16x2048 f16 probability strip held in LDS (64KB/wave, 256KB/WGP)
//  - logits computed once (QK^T via v_wmma_f32_16x16x32_f16), weights
//    normalized on the fly and written exactly once with b128 stores
//  - softmax sweep and weight streaming fully vectorized (b64 LDS ops)
// ---------------------------------------------------------------------------

typedef __attribute__((ext_vector_type(16))) _Float16 v16h;
typedef __attribute__((ext_vector_type(8)))  _Float16 v8h;
typedef __attribute__((ext_vector_type(4)))  _Float16 v4h;
typedef __attribute__((ext_vector_type(8)))  float    v8f;
typedef __attribute__((ext_vector_type(4)))  float    f32x4;

#define BB 4
#define HH 16
#define SS 2048
#define DD 128

#define WAVES_PER_BLOCK 4            // each wave owns a 16-row query strip
#define QROWS_PER_BLOCK (WAVES_PER_BLOCK * 16)
#define PBUF_BYTES (WAVES_PER_BLOCK * 16 * SS * 2)        // 256 KB f16 strips
#define SMEM_BYTES (PBUF_BYTES + 2 * WAVES_PER_BLOCK * 16 * 4)  // + max/inv

__global__ __launch_bounds__(WAVES_PER_BLOCK * 32)
void sdpa_wmma_kernel(const float* __restrict__ Q,
                      const float* __restrict__ Kt,
                      const float* __restrict__ V,
                      const int*   __restrict__ Msk,
                      float* __restrict__ Out,     // [B,H,S,D]
                      float* __restrict__ Wts)     // [B,H,S,S]
{
    extern __shared__ char smem[];

    const int lane = threadIdx.x & 31;
    const int wv   = threadIdx.x >> 5;
    const int n    = lane & 15;        // WMMA column within tile
    const int half = lane >> 4;        // 0: lanes 0-15, 1: lanes 16-31

    const int bh    = blockIdx.x >> 5;           // 0..63  (B*H)
    const int qblk  = blockIdx.x & 31;           // 0..31  (S / 64)
    const int b     = bh >> 4;                   // batch (H==16)
    const int qbase = qblk * QROWS_PER_BLOCK + wv * 16;

    _Float16* pbuf   = (_Float16*)(smem) + (size_t)wv * 16 * SS;   // [16][SS]
    float*    rowmax = (float*)(smem + PBUF_BYTES) + wv * 16;
    float*    rowinv = (float*)(smem + PBUF_BYTES + WAVES_PER_BLOCK * 16 * 4)
                       + wv * 16;

    const float scale = 0.08838834764831845f;    // 1/sqrt(128)
    const float LOG2E = 1.4426950408889634f;

    // ------------------------------------------------------------------
    // Q A-fragments: 4 fragments of 16x32 f16 covering D=128, pre-scaled.
    // A layout (16-bit, 16x32): lane m = lane&15; kb = (lane>>4)*16;
    //   elements 0..15 -> K = kb + e
    // ------------------------------------------------------------------
    v16h qa[4];
    {
        const int    row  = qbase + n;
        const float* qrow = Q + ((size_t)bh * SS + row) * DD;
        const int    kb   = half * 16;
#pragma unroll
        for (int c = 0; c < 4; c++) {
            const float* p0 = qrow + c * 32 + kb;
            const float* p1 = qrow + c * 32 + 8 + kb;
            f32x4 x0 = *(const f32x4*)(p0);
            f32x4 x1 = *(const f32x4*)(p0 + 4);
            f32x4 y0 = *(const f32x4*)(p1);
            f32x4 y1 = *(const f32x4*)(p1 + 4);
            v16h a;
#pragma unroll
            for (int e = 0; e < 4; e++) {
                a[e]      = (_Float16)(x0[e] * scale);
                a[4 + e]  = (_Float16)(x1[e] * scale);
                a[8 + e]  = (_Float16)(y0[e] * scale);
                a[12 + e] = (_Float16)(y1[e] * scale);
            }
            qa[c] = a;
        }
    }

    // ------------------------------------------------------------------
    // Pass 1: logits = (Q/sqrt(d)) . K^T over all keys, masked; per-lane
    // running row max; masked logits stashed as f16 into the LDS strip.
    // C layout: lane col N = lane&15; acc[j] row = j + 8*half.
    // ------------------------------------------------------------------
    float mrun[8];
#pragma unroll
    for (int j = 0; j < 8; j++) mrun[j] = -3.0e38f;

    const float* kbptr = Kt + (size_t)bh * SS * DD;
    const int*   mbase = Msk + ((size_t)b * SS + qbase) * SS;

    for (int t = 0; t < SS / 16; t++) {
        const int    keyrow = t * 16 + n;
        const float* krow   = kbptr + (size_t)keyrow * DD;
        const int    kb16   = half * 16;

        // prefetch next key tile row (global_prefetch_b8)
        if (t + 1 < SS / 16) __builtin_prefetch(krow + 16 * DD, 0, 1);

        v8f acc = {};
#pragma unroll
        for (int c = 0; c < 4; c++) {
            // B layout (16-bit, 32x16): lane col = lane&15; K = kb16 + e
            const float* p0 = krow + c * 32 + kb16;
            f32x4 x0 = *(const f32x4*)(p0);
            f32x4 x1 = *(const f32x4*)(p0 + 4);
            f32x4 x2 = *(const f32x4*)(p0 + 8);
            f32x4 x3 = *(const f32x4*)(p0 + 12);
            v16h bf;
#pragma unroll
            for (int e = 0; e < 4; e++) {
                bf[e]      = (_Float16)x0[e];
                bf[4 + e]  = (_Float16)x1[e];
                bf[8 + e]  = (_Float16)x2[e];
                bf[12 + e] = (_Float16)x3[e];
            }
            acc = __builtin_amdgcn_wmma_f32_16x16x32_f16(
                false, qa[c], false, bf, (short)0, acc, false, false);
        }

#pragma unroll
        for (int j = 0; j < 8; j++) {
            const int mloc = j + half * 8;                    // local q row
            const int mk   = mbase[(size_t)mloc * SS + t * 16 + n];
            const float x  = (mk == 0) ? -3.0e38f : acc[j];   // f16 -> -inf
            mrun[j] = fmaxf(mrun[j], x);
            pbuf[(size_t)mloc * SS + t * 16 + n] = (_Float16)x;
        }
    }

    // row max: butterfly across the 16-lane half (C rows live per-half)
#pragma unroll
    for (int j = 0; j < 8; j++) {
#pragma unroll
        for (int off = 8; off >= 1; off >>= 1)
            mrun[j] = fmaxf(mrun[j], __shfl_xor(mrun[j], off, 32));
    }
    if (n == 0) {
#pragma unroll
        for (int j = 0; j < 8; j++) rowmax[j + half * 8] = mrun[j];
    }

    // ------------------------------------------------------------------
    // Pass 2 (row-major, vectorized): p = exp2((l - m) * log2e); row sums;
    // overwrite the strip with unnormalized probabilities. b64 LDS ops.
    // ------------------------------------------------------------------
    for (int r = 0; r < 16; r++) {
        const float m  = rowmax[r];              // LDS broadcast (same wave)
        _Float16*   pr = pbuf + (size_t)r * SS;
        float sum = 0.0f;
#pragma unroll 4
        for (int i = 0; i < SS / 128; i++) {
            const int key = i * 128 + lane * 4;
            v4h h = *(v4h*)(pr + key);
            f32x4 p;
#pragma unroll
            for (int e = 0; e < 4; e++) {
                p[e] = exp2f(((float)h[e] - m) * LOG2E);
                sum += p[e];
                h[e] = (_Float16)p[e];
            }
            *(v4h*)(pr + key) = h;
        }
#pragma unroll
        for (int off = 16; off >= 1; off >>= 1)
            sum += __shfl_xor(sum, off, 32);
        if (lane == 0) rowinv[r] = 1.0f / sum;
    }

    // ------------------------------------------------------------------
    // Pass 3: O = P . V  (P f16 from LDS as A-fragments via ds_load_b128;
    // V gathered/converted to f16 B-fragments). 8 acc tiles cover D=128.
    // ------------------------------------------------------------------
    v8f oacc[8];
#pragma unroll
    for (int dt = 0; dt < 8; dt++)
#pragma unroll
        for (int e = 0; e < 8; e++) oacc[dt][e] = 0.0f;

    const float* vbase = V + (size_t)bh * SS * DD;
    const int    kbA   = half * 16;
    const int    kb16  = half * 16;

    for (int kt = 0; kt < SS / 32; kt++) {
        // A fragment: 2x ds_load_b128 from the probability strip
        const _Float16* ap = pbuf + (size_t)n * SS + kt * 32 + kbA;
        v8h lo = *(const v8h*)(ap);
        v8h hi = *(const v8h*)(ap + 8);
        v16h pa;
#pragma unroll
        for (int e = 0; e < 8; e++) { pa[e] = lo[e]; pa[8 + e] = hi[e]; }

#pragma unroll
        for (int dt = 0; dt < 8; dt++) {
            const float* vp = vbase + (size_t)(kt * 32 + kb16) * DD + dt * 16 + n;
            v16h bf;
#pragma unroll
            for (int e = 0; e < 16; e++) bf[e] = (_Float16)vp[(size_t)e * DD];
            oacc[dt] = __builtin_amdgcn_wmma_f32_16x16x32_f16(
                false, pa, false, bf, (short)0, oacc[dt], false, false);
        }
    }

    // write attn_output, normalized (inv per C row from rowinv)
    float invj[8];
#pragma unroll
    for (int j = 0; j < 8; j++) invj[j] = rowinv[j + half * 8];

    float* obase = Out + ((size_t)bh * SS + qbase) * DD;
#pragma unroll
    for (int dt = 0; dt < 8; dt++) {
#pragma unroll
        for (int j = 0; j < 8; j++) {
            const int mloc = j + half * 8;
            obase[(size_t)mloc * DD + dt * 16 + n] = oacc[dt][j] * invj[j];
        }
    }

    // ------------------------------------------------------------------
    // Pass 4: stream normalized weights to global, b64 LDS read +
    // b128 coalesced global store per lane-iteration.
    // ------------------------------------------------------------------
    float* wbase = Wts + ((size_t)bh * SS + qbase) * SS;
    for (int r = 0; r < 16; r++) {
        const float      iv = rowinv[r];          // LDS broadcast
        const _Float16*  pr = pbuf + (size_t)r * SS;
        float*           wr = wbase + (size_t)r * SS;
#pragma unroll 4
        for (int kc = 0; kc < SS / 128; kc++) {
            const int key = kc * 128 + lane * 4;
            v4h h = *(const v4h*)(pr + key);
            f32x4 w;
#pragma unroll
            for (int e = 0; e < 4; e++) w[e] = (float)h[e] * iv;
            *(f32x4*)(wr + key) = w;
        }
    }
}

extern "C" void kernel_launch(void* const* d_in, const int* in_sizes, int n_in,
                              void* d_out, int out_size, void* d_ws, size_t ws_size,
                              hipStream_t stream) {
    (void)in_sizes; (void)n_in; (void)out_size; (void)d_ws; (void)ws_size;
    const float* Q   = (const float*)d_in[0];
    const float* Kt  = (const float*)d_in[1];
    const float* V   = (const float*)d_in[2];
    const int*   Msk = (const int*)d_in[3];

    float* Out = (float*)d_out;                               // [B,H,S,D]
    float* Wts = Out + (size_t)BB * HH * SS * DD;              // [B,H,S,S]

    const size_t shmem = SMEM_BYTES;                           // 262656 B
    hipFuncSetAttribute((const void*)sdpa_wmma_kernel,
                        hipFuncAttributeMaxDynamicSharedMemorySize,
                        (int)shmem);

    const int blocks = BB * HH * (SS / QROWS_PER_BLOCK);       // 2048
    sdpa_wmma_kernel<<<dim3(blocks), dim3(WAVES_PER_BLOCK * 32), shmem, stream>>>(
        Q, Kt, V, Msk, Out, Wts);
}
